// GPT_63445256896952
// MI455X (gfx1250) — matrix operand order repaired and checked by reading, not compiled, not run on hardware
//
#include <hip/hip_runtime.h>
#include <hip/hip_bf16.h>
#include <math.h>

// ---------------- model constants ----------------
#define B_   4
#define T_   512
#define L_   12
#define H_   12
#define G_   4
#define D_   768
#define HD_  64
#define KV_  256
#define WIN_ 128
#define V_   50304
#define ROWS (B_ * T_)   // 2048 token rows

// Use CDNA5 async global->LDS DMA for the (bf16) A-tile staging.
#define USE_ASYNC_A 1

// ---------------- WMMA types ----------------
typedef __bf16 bf16_t;
typedef bf16_t bf16x16 __attribute__((ext_vector_type(16)));
typedef float  f32x8  __attribute__((ext_vector_type(8)));

// round-to-nearest-even fp32->bf16, two packed into a dword
__device__ __forceinline__ unsigned int pack_bf16(float a, float b) {
  unsigned int ua = __float_as_uint(a);
  unsigned int ub = __float_as_uint(b);
  ua += 0x7fffu + ((ua >> 16) & 1u);
  ub += 0x7fffu + ((ub >> 16) & 1u);
  return (ua >> 16) | (ub & 0xffff0000u);
}
__device__ __forceinline__ unsigned short f2bf(float a) {
  unsigned int ua = __float_as_uint(a);
  ua += 0x7fffu + ((ua >> 16) & 1u);
  return (unsigned short)(ua >> 16);
}

// ---------------- embedding: x = wte[idx] + wpe[t] (fp32 residual) ----------------
__global__ __launch_bounds__(256)
void embed_k(const int* __restrict__ idx, const float* __restrict__ wte,
             const float* __restrict__ wpe, float* __restrict__ x) {
  int t4  = blockIdx.x * blockDim.x + threadIdx.x;   // ROWS*192 float4 slots
  int row = t4 / (D_ / 4);
  int c   = (t4 % (D_ / 4)) * 4;
  int tok = idx[row];
  int tp  = row % T_;
  float4 a = *(const float4*)(wte + (size_t)tok * D_ + c);
  float4 p = *(const float4*)(wpe + (size_t)tp  * D_ + c);
  a.x += p.x; a.y += p.y; a.z += p.z; a.w += p.w;
  *(float4*)(x + (size_t)row * D_ + c) = a;
}

// ---------------- LayerNorm -> packed bf16, one wave32 per row ----------------
__global__ __launch_bounds__(256)
void ln_rows(const float* __restrict__ x, const float* __restrict__ w,
             unsigned short* __restrict__ out) {
  int wave = (blockIdx.x * blockDim.x + threadIdx.x) >> 5;  // row id
  int lane = threadIdx.x & 31;
  const float* xr = x + (size_t)wave * D_;
  float4 vals[6];
  float s = 0.f, sq = 0.f;
  for (int i = 0; i < 6; ++i) {
    vals[i] = *(const float4*)(xr + i * 128 + lane * 4);
    s  += vals[i].x + vals[i].y + vals[i].z + vals[i].w;
    sq += vals[i].x * vals[i].x + vals[i].y * vals[i].y +
          vals[i].z * vals[i].z + vals[i].w * vals[i].w;
  }
  for (int off = 16; off > 0; off >>= 1) {
    s  += __shfl_xor(s,  off, 32);
    sq += __shfl_xor(sq, off, 32);
  }
  float mu   = s  * (1.0f / (float)D_);
  float var  = sq * (1.0f / (float)D_) - mu * mu;
  float rstd = rsqrtf(var + 1e-5f);
  unsigned short* orow = out + (size_t)wave * D_;
  for (int i = 0; i < 6; ++i) {
    int d = i * 128 + lane * 4;
    float4 wv = *(const float4*)(w + d);
    uint2 pv;
    pv.x = pack_bf16((vals[i].x - mu) * rstd * wv.x, (vals[i].y - mu) * rstd * wv.y);
    pv.y = pack_bf16((vals[i].z - mu) * rstd * wv.z, (vals[i].w - mu) * rstd * wv.w);
    *(uint2*)(orow + d) = pv;
  }
}

// ---------------- WMMA GEMM: C[M,N] = A_bf16[M,K] * B_f32[K,N] ----------------
// A is bf16 in global; staged to LDS via CDNA5 async DMA (ASYNCcnt-tracked).
// B is fp32 weights; packed-converted to bf16 in the register-commit pipeline.
// Block tile 128x128, 8 waves; each wave owns 16x128 (8 accumulators): one
// A-fragment feeds 8 WMMAs per K-step. One barrier per K-step.
// EPI: 0 = store fp32, 1 = fp32 +=, 2 = exact GELU -> bf16 store
// BT : false -> B row-major [K,N]; true -> B given as [N,K] (lm_head: wte^T)
#define BM 128
#define BN 128
#define BK 32
#define LDA_S 40
#define LDB_S 40

template<int EPI, bool BT, int N, int K>
__global__ __launch_bounds__(256)
void gemm_bf16w(const unsigned short* __restrict__ Abf,
                const float* __restrict__ Bsrc, void* __restrict__ Cv) {
  __shared__ __align__(16) unsigned short sA[2][BM * LDA_S];   // [128][32] bf16 (+pad)
  __shared__ __align__(16) unsigned short sB[2][BN * LDB_S];   // transposed: [n][k]

  const int tid  = threadIdx.x;
  const int lane = tid & 31;
  const int wid  = tid >> 5;         // 8 waves along M
  const int m0   = blockIdx.y * BM;
  const int n0   = blockIdx.x * BN;
  const int half = lane >> 4;
  const int lr   = lane & 15;

  // per-thread staging coordinates
  const int ar = tid >> 1;                 // A row 0..127
  const int ac = (tid & 1) * 16;           // A col 0 or 16 (16 bf16 = 32B)
  const int bk = (tid >> 4) * 2;           // !BT: k pair 0,2,..,30
  const int bn = (tid & 15) * 8;           // !BT: n 0..120
  const int tn = tid >> 1;                 // BT: n 0..127
  const int tc = (tid & 1) * 16;           // BT: k off 0 or 16

  f32x8 acc[8];
  for (int j = 0; j < 8; ++j)
    for (int e = 0; e < 8; ++e) acc[j][e] = 0.0f;

  float4 rb0, rb1, rb2, rb3;
#if !USE_ASYNC_A
  uint4 raw0, raw1;
#endif

  // ---- prologue: tile 0 ----
  {
    const unsigned short* ap = Abf + (size_t)(m0 + ar) * K + ac;
#if USE_ASYNC_A
    unsigned laddr = (unsigned)(size_t)&sA[0][ar * LDA_S + ac];
    unsigned long long gaddr = (unsigned long long)(size_t)ap;
    asm volatile("global_load_async_to_lds_b128 %0, %1, off"
                 :: "v"(laddr), "v"(gaddr) : "memory");
    asm volatile("global_load_async_to_lds_b128 %0, %1, off offset:16"
                 :: "v"(laddr), "v"(gaddr) : "memory");
#else
    raw0 = *(const uint4*)(ap);
    raw1 = *(const uint4*)(ap + 8);
#endif
    if (!BT) {
      const float* bp = Bsrc + (size_t)bk * N + n0 + bn;
      rb0 = *(const float4*)(bp);
      rb1 = *(const float4*)(bp + 4);
      rb2 = *(const float4*)(bp + N);
      rb3 = *(const float4*)(bp + N + 4);
    } else {
      const float* bp = Bsrc + (size_t)(n0 + tn) * K + tc;
      rb0 = *(const float4*)(bp);
      rb1 = *(const float4*)(bp + 4);
      rb2 = *(const float4*)(bp + 8);
      rb3 = *(const float4*)(bp + 12);
    }
  }

  constexpr int NK = K / BK;
  for (int kt = 0; kt < NK; ++kt) {
    const int buf = kt & 1;
    // ---- commit B registers (convert + pack) into sB[buf] ----
    {
      if (!BT) {
        unsigned short* base = &sB[buf][0];
        *(unsigned int*)&base[(bn + 0) * LDB_S + bk] = pack_bf16(rb0.x, rb2.x);
        *(unsigned int*)&base[(bn + 1) * LDB_S + bk] = pack_bf16(rb0.y, rb2.y);
        *(unsigned int*)&base[(bn + 2) * LDB_S + bk] = pack_bf16(rb0.z, rb2.z);
        *(unsigned int*)&base[(bn + 3) * LDB_S + bk] = pack_bf16(rb0.w, rb2.w);
        *(unsigned int*)&base[(bn + 4) * LDB_S + bk] = pack_bf16(rb1.x, rb3.x);
        *(unsigned int*)&base[(bn + 5) * LDB_S + bk] = pack_bf16(rb1.y, rb3.y);
        *(unsigned int*)&base[(bn + 6) * LDB_S + bk] = pack_bf16(rb1.z, rb3.z);
        *(unsigned int*)&base[(bn + 7) * LDB_S + bk] = pack_bf16(rb1.w, rb3.w);
      } else {
        uint4 q0, q1;
        q0.x = pack_bf16(rb0.x, rb0.y); q0.y = pack_bf16(rb0.z, rb0.w);
        q0.z = pack_bf16(rb1.x, rb1.y); q0.w = pack_bf16(rb1.z, rb1.w);
        q1.x = pack_bf16(rb2.x, rb2.y); q1.y = pack_bf16(rb2.z, rb2.w);
        q1.z = pack_bf16(rb3.x, rb3.y); q1.w = pack_bf16(rb3.z, rb3.w);
        *(uint4*)&sB[buf][tn * LDB_S + tc]     = q0;
        *(uint4*)&sB[buf][tn * LDB_S + tc + 8] = q1;
      }
#if !USE_ASYNC_A
      *(uint4*)&sA[buf][ar * LDA_S + ac]     = raw0;
      *(uint4*)&sA[buf][ar * LDA_S + ac + 8] = raw1;
#endif
    }
#if USE_ASYNC_A
    // own async DMA for tile kt must be LDS-resident before the barrier
    asm volatile("s_wait_asynccnt 0x0" ::: "memory");
#endif
    __syncthreads();   // all waves' DMA + sB commits now visible

    // ---- issue loads for tile kt+1 (overlap with WMMAs below) ----
    if (kt + 1 < NK) {
      const int k0n = (kt + 1) * BK;
      const unsigned short* ap = Abf + (size_t)(m0 + ar) * K + k0n + ac;
#if USE_ASYNC_A
      unsigned laddr = (unsigned)(size_t)&sA[buf ^ 1][ar * LDA_S + ac];
      unsigned long long gaddr = (unsigned long long)(size_t)ap;
      asm volatile("global_load_async_to_lds_b128 %0, %1, off"
                   :: "v"(laddr), "v"(gaddr) : "memory");
      asm volatile("global_load_async_to_lds_b128 %0, %1, off offset:16"
                   :: "v"(laddr), "v"(gaddr) : "memory");
#else
      raw0 = *(const uint4*)(ap);
      raw1 = *(const uint4*)(ap + 8);
#endif
      if (!BT) {
        const float* bp = Bsrc + (size_t)(k0n + bk) * N + n0 + bn;
        rb0 = *(const float4*)(bp);
        rb1 = *(const float4*)(bp + 4);
        rb2 = *(const float4*)(bp + N);
        rb3 = *(const float4*)(bp + N + 4);
      } else {
        const float* bp = Bsrc + (size_t)(n0 + tn) * K + k0n + tc;
        rb0 = *(const float4*)(bp);
        rb1 = *(const float4*)(bp + 4);
        rb2 = *(const float4*)(bp + 8);
        rb3 = *(const float4*)(bp + 12);
      }
    }

    // ---- one A fragment feeds 8 WMMAs on buffer `buf` ----
    union FragU { uint4 u[2]; bf16x16 v; };
    FragU af;
    const unsigned short* ap = &sA[buf][(wid * 16 + lr) * LDA_S + half * 16];
    af.u[0] = *(const uint4*)(ap);        // K = half*16 + [0..7]
    af.u[1] = *(const uint4*)(ap + 8);    // K = half*16 + [8..15]
    for (int j = 0; j < 8; ++j) {
      FragU bfm;
      const unsigned short* bp = &sB[buf][(j * 16 + lr) * LDB_S + half * 16];
      bfm.u[0] = *(const uint4*)(bp);      // K = half*16 + [0..7]
      bfm.u[1] = *(const uint4*)(bp + 8);  // K = half*16 + [8..15]
      acc[j] = __builtin_amdgcn_wmma_f32_16x16x32_bf16(
          false, af.v, false, bfm.v, (short)0, acc[j], false, false);
    }
  }

  // ---- epilogue (C/D layout: row = r + 8*half, col = lane&15) ----
  for (int j = 0; j < 8; ++j) {
    int gn = n0 + j * 16 + lr;
    for (int r = 0; r < 8; ++r) {
      int gm = m0 + wid * 16 + half * 8 + r;
      size_t o = (size_t)gm * N + gn;
      float vv = acc[j][r];
      if (EPI == 0) {
        ((float*)Cv)[o] = vv;
      } else if (EPI == 1) {
        ((float*)Cv)[o] += vv;
      } else {
        float g = 0.5f * vv * (1.0f + erff(vv * 0.70710678118654752f));  // exact GELU
        ((unsigned short*)Cv)[o] = f2bf(g);
      }
    }
  }
}

// ---------------- attention: one wave32 per (b,h,q); fp32 math, bf16 output ------------
__global__ __launch_bounds__(256)
void attn_k(const float* __restrict__ q, const float* __restrict__ k,
            const float* __restrict__ v, unsigned short* __restrict__ y) {
  __shared__ float qv[8][64];
  __shared__ float sc[8][136];
  int w    = threadIdx.x >> 5;
  int lane = threadIdx.x & 31;
  int gw   = blockIdx.x * 8 + w;
  int qpos = gw % T_;
  int h    = (gw / T_) % H_;
  int b    = gw / (T_ * H_);
  int g    = h / (H_ / G_);           // GQA repeat_interleave group

  const float* qrow = q + (size_t)(b * T_ + qpos) * D_ + h * HD_;
  *(float2*)&qv[w][lane * 2] = *(const float2*)(qrow + lane * 2);
  __syncthreads();

  int jstart = qpos - WIN_; if (jstart < 0) jstart = 0;
  int nj = qpos - jstart + 1;         // 1..129

  for (int jj = lane; jj < nj; jj += 32) {
    const float* krow = k + (size_t)(b * T_ + jstart + jj) * KV_ + g * HD_;
    float dot = 0.f;
    for (int d = 0; d < HD_; ++d) dot += qv[w][d] * krow[d];
    sc[w][jj] = dot * 0.125f;         // 1/sqrt(64)
  }
  __syncthreads();

  float mx = -3.4e38f;
  for (int jj = lane; jj < nj; jj += 32) mx = fmaxf(mx, sc[w][jj]);
  for (int off = 16; off > 0; off >>= 1) mx = fmaxf(mx, __shfl_xor(mx, off, 32));
  float sum = 0.f;
  for (int jj = lane; jj < nj; jj += 32) {
    float e = __expf(sc[w][jj] - mx);
    sc[w][jj] = e;
    sum += e;
  }
  for (int off = 16; off > 0; off >>= 1) sum += __shfl_xor(sum, off, 32);
  float inv = 1.f / sum;
  __syncthreads();

  float a0 = 0.f, a1 = 0.f;
  for (int jj = 0; jj < nj; ++jj) {
    float p = sc[w][jj];
    const float* vrow = v + (size_t)(b * T_ + jstart + jj) * KV_ + g * HD_ + lane * 2;
    float2 vv = *(const float2*)vrow;
    a0 += p * vv.x; a1 += p * vv.y;
  }
  *(unsigned int*)(y + (size_t)(b * T_ + qpos) * D_ + h * HD_ + lane * 2) =
      pack_bf16(a0 * inv, a1 * inv);
}

// ---------------- host launcher ----------------
extern "C" void kernel_launch(void* const* d_in, const int* in_sizes, int n_in,
                              void* d_out, int out_size, void* d_ws, size_t ws_size,
                              hipStream_t stream) {
  (void)in_sizes; (void)n_in; (void)out_size; (void)ws_size;
  const int*   idx = (const int*)  d_in[0];
  const float* wte = (const float*)d_in[1];
  const float* wpe = (const float*)d_in[2];
  const float* ln1 = (const float*)d_in[3];
  const float* ln2 = (const float*)d_in[4];
  const float* lnf = (const float*)d_in[5];
  const float* Wq  = (const float*)d_in[6];
  const float* Wk  = (const float*)d_in[7];
  const float* Wv  = (const float*)d_in[8];
  const float* Wo  = (const float*)d_in[9];
  const float* Wfc = (const float*)d_in[10];
  const float* Wpr = (const float*)d_in[11];
  float* out = (float*)d_out;

  // workspace layout: fp32 residual + qkv, bf16 GEMM-A activations (~36 MB)
  float*          x   = (float*)d_ws;
  float*          qb  = x  + (size_t)ROWS * D_;
  float*          kb  = qb + (size_t)ROWS * D_;
  float*          vb  = kb + (size_t)ROWS * KV_;
  unsigned short* xln = (unsigned short*)(vb + (size_t)ROWS * KV_);
  unsigned short* yb  = xln + (size_t)ROWS * D_;
  unsigned short* hb  = yb  + (size_t)ROWS * D_;   // [ROWS, 4*D] bf16

  embed_k<<<ROWS * (D_ / 4) / 256, 256, 0, stream>>>(idx, wte, wpe, x);

  dim3 blk(256);
  dim3 gD (D_ / BN,      ROWS / BM);      // (6, 16)
  dim3 gKV(KV_ / BN,     ROWS / BM);      // (2, 16)
  dim3 gFC(4 * D_ / BN,  ROWS / BM);      // (24, 16)
  dim3 gLM(V_ / BN,      ROWS / BM);      // (393, 16)

  for (int l = 0; l < L_; ++l) {
    ln_rows<<<ROWS / 8, blk, 0, stream>>>(x, ln1 + (size_t)l * D_, xln);
    gemm_bf16w<0, false, D_,     D_    ><<<gD,  blk, 0, stream>>>(xln, Wq  + (size_t)l * D_ * D_,     qb);
    gemm_bf16w<0, false, KV_,    D_    ><<<gKV, blk, 0, stream>>>(xln, Wk  + (size_t)l * D_ * KV_,    kb);
    gemm_bf16w<0, false, KV_,    D_    ><<<gKV, blk, 0, stream>>>(xln, Wv  + (size_t)l * D_ * KV_,    vb);
    attn_k<<<(B_ * H_ * T_) / 8, blk, 0, stream>>>(qb, kb, vb, yb);
    gemm_bf16w<1, false, D_,     D_    ><<<gD,  blk, 0, stream>>>(yb,  Wo  + (size_t)l * D_ * D_,     x);
    ln_rows<<<ROWS / 8, blk, 0, stream>>>(x, ln2 + (size_t)l * D_, xln);
    gemm_bf16w<2, false, 4 * D_, D_    ><<<gFC, blk, 0, stream>>>(xln, Wfc + (size_t)l * D_ * 4 * D_, hb);
    gemm_bf16w<1, false, D_,     4 * D_><<<gD,  blk, 0, stream>>>(hb,  Wpr + (size_t)l * 4 * D_ * D_, x);
  }
  ln_rows<<<ROWS / 8, blk, 0, stream>>>(x, lnf, xln);
  // tied lm_head: out = xln @ wte^T  (BT=true: B indexed [N,K], contiguous along K)
  gemm_bf16w<0, true, V_, D_><<<gLM, blk, 0, stream>>>(xln, wte, out);
}
